// VisualWordTokenizer_38371237822605
// MI455X (gfx1250) — matrix operand order repaired and checked by reading, not compiled, not run on hardware
//
#include <hip/hip_runtime.h>
#include <hip/hip_bf16.h>

// ---------------------------------------------------------------------------
// ViT-B/16 visual-word tokenizer for MI455X (gfx1250, wave32, WMMA + TDM)
//   pixel_values [256,3,224,224] f32 -> out [256,99,768] f32
// One pixel pass (tokenize f16 + variance score), top-98 select BEFORE the
// embedding GEMM, then a 112x768x768 f16 WMMA GEMM per image with
// double-buffered LDS; B tiles staged by the Tensor Data Mover.
// ---------------------------------------------------------------------------

#define VWT_USE_TDM 1

typedef __attribute__((ext_vector_type(16))) _Float16 v16h;
typedef __attribute__((ext_vector_type(8)))  _Float16 v8h;
typedef __attribute__((ext_vector_type(8)))  float    v8f;
typedef __attribute__((ext_vector_type(4)))  unsigned int u32x4;
typedef __attribute__((ext_vector_type(8)))  unsigned int u32x8;

#define VWT_B       256
#define VWT_HIDDEN  768
#define VWT_NPATCH  196
#define VWT_GRID    14
#define VWT_TOPK    98
#define VWT_ROWS    112      // 98 padded to 7 WMMA row tiles
#define VWT_LDA     40       // sA row stride in halves (80B, 16B aligned, bank-tiling)
#define VWT_LDB     136      // sBt row stride in halves (272B = 256B row + 16B pad)

// ---------------------------------------------------------------------------
// Kernel 0: transpose + cast conv_w [768(d),768(k)] f32 -> wT [768(k),768(d)] f16
// ---------------------------------------------------------------------------
__global__ void vwt_wprep(const float* __restrict__ w, _Float16* __restrict__ wT) {
    int idx = blockIdx.x * 256 + threadIdx.x;      // 589824 total, writes coalesced
    int k = idx / VWT_HIDDEN;
    int d = idx - k * VWT_HIDDEN;
    wT[idx] = (_Float16)w[d * VWT_HIDDEN + k];
}

// ---------------------------------------------------------------------------
// Kernel 1: one wave32 per patch; single pixel pass -> f16 token row + variance
// ---------------------------------------------------------------------------
__global__ __launch_bounds__(256) void vwt_score_token(
        const float* __restrict__ pix,
        _Float16* __restrict__ tokens,
        float* __restrict__ scores) {
    int wid  = (blockIdx.x * 256 + threadIdx.x) >> 5;   // global wave id = patch id
    int lane = threadIdx.x & 31;
    int b = wid / VWT_NPATCH;
    int p = wid - b * VWT_NPATCH;
    int gh = p / VWT_GRID;
    int gw = p - gh * VWT_GRID;

    const float* base = pix + (size_t)b * 3 * 224 * 224;
    size_t tokbase = (size_t)wid * VWT_HIDDEN;

    float sum = 0.0f, sumsq = 0.0f;
#pragma unroll
    for (int i = 0; i < 24; ++i) {
        int kk = lane + (i << 5);
        int c  = kk >> 8;
        int ph = (kk >> 4) & 15;
        int pw = kk & 15;
        float v = base[((size_t)c * 224 + gh * 16 + ph) * 224 + gw * 16 + pw];
        sum   += v;
        sumsq += v * v;
        tokens[tokbase + kk] = (_Float16)v;
    }
#pragma unroll
    for (int off = 16; off > 0; off >>= 1) {
        sum   += __shfl_xor(sum,   off, 32);
        sumsq += __shfl_xor(sumsq, off, 32);
    }
    if (lane == 0) {
        // jnp.var(ddof=1): (sumsq - sum^2/n) / (n-1)
        scores[wid] = (sumsq - sum * sum * (1.0f / 768.0f)) * (1.0f / 767.0f);
    }
}

// ---------------------------------------------------------------------------
// Kernel 2: per-image top-98 rank select (ties -> lower idx), ascending order
// ---------------------------------------------------------------------------
__global__ __launch_bounds__(256) void vwt_topk(
        const float* __restrict__ scores, int* __restrict__ sel) {
    int b = blockIdx.x;
    int t = threadIdx.x;
    __shared__ float s[VWT_NPATCH];
    __shared__ int   flag[VWT_NPATCH];

    if (t < VWT_NPATCH) s[t] = scores[b * VWT_NPATCH + t];
    __syncthreads();
    if (t < VWT_NPATCH) {
        float st = s[t];
        int rank = 0;
        for (int j = 0; j < VWT_NPATCH; ++j) {
            float sj = s[j];
            rank += (sj > st) || (sj == st && j < t);
        }
        flag[t] = (rank < VWT_TOPK) ? 1 : 0;
    }
    __syncthreads();
    if (t < VWT_NPATCH && flag[t]) {
        int pos = 0;
        for (int j = 0; j < t; ++j) pos += flag[j];
        sel[b * VWT_TOPK + pos] = t;
    }
}

// ---------------------------------------------------------------------------
// Kernel 3: CLS row = cls_token + pos_emb[0]
// ---------------------------------------------------------------------------
__global__ void vwt_cls_row(const float* __restrict__ cls,
                            const float* __restrict__ pos,
                            float* __restrict__ out) {
    int b = blockIdx.x;
    int d = blockIdx.y * 256 + threadIdx.x;            // 0..767
    out[(size_t)b * 99 * VWT_HIDDEN + d] = cls[d] + pos[d];
}

#if VWT_USE_TDM
// ---------------------------------------------------------------------------
// Issue TENSOR_LOAD_TO_LDS for one B tile: 32 rows x 128 f16 from wT
// (row stride 768 halves) into LDS with 272B row stride (TDM pad: +4 dwords
// after every 64 dwords). D# per CDNA5 ISA.
// Wave-level op (EXEC ignored); issue from one wave only.
// ---------------------------------------------------------------------------
__device__ __forceinline__ void vwt_tdm_load_b(const _Float16* gsrc, _Float16* ldst) {
    unsigned long long ga = (unsigned long long)(uintptr_t)gsrc;
    unsigned int       la = (unsigned int)(uintptr_t)ldst;     // low 32 = LDS offset
    u32x4 g0;
    g0[0] = 1u;                                   // count=1, user descriptor
    g0[1] = la;                                   // lds_addr
    g0[2] = (unsigned int)ga;                     // global_addr[31:0]
    g0[3] = (unsigned int)((ga >> 32) & 0x01FFFFFFu) | 0x80000000u; // [56:32] | type=2
    u32x8 g1;
    g1[0] = 0x00010000u                           // data_size = 1 (2 bytes)
          | 0x00100000u                           // pad_enable
          | (5u << 22)                            // pad_interval: 64 dwords (256B row)
          | (3u << 25);                           // pad_amount : 4 dwords (16B)
    g1[1] = ((unsigned)VWT_HIDDEN & 0xFFFFu) << 16; // tensor_dim0[15:0] = 768
    g1[2] = ((unsigned)VWT_HIDDEN & 0xFFFFu) << 16; // dim0 hi=0 | tensor_dim1[15:0]=768
    g1[3] = (128u << 16);                         // dim1 hi=0 | tile_dim0 = 128
    g1[4] = 32u;                                  // tile_dim1 = 32, tile_dim2 = 0
    g1[5] = (unsigned)VWT_HIDDEN;                 // tensor_dim0_stride = 768
    g1[6] = 0u;
    g1[7] = 0u;
    asm volatile("tensor_load_to_lds %0, %1" :: "s"(g0), "s"(g1) : "memory");
}
#endif

// ---------------------------------------------------------------------------
// Kernel 4: WMMA GEMM over the 98 selected patches (padded to 112 rows).
// grid = (256 images, 6 col-blocks of 128); 256 threads = 8 waves; wave owns
// one 16-col strip x 7 row tiles. Double-buffered LDS, one barrier per K-step.
// Fragment layouts per CDNA5 ISA 7.12.2:
//   A 16x32 f16 : lane -> M = lane%16; two 8-half K runs (16*(e/8)+8*half+e%8)
//   B 32x16 f16 : lane -> N = lane%16; element e -> K as for A (16*(e/8)+8*half+e%8)
//   C/D 16x16 f32: VGPR r -> M = r + 8*half; N = lane%16
// ---------------------------------------------------------------------------
__global__ __launch_bounds__(256) void vwt_gemm(
        const _Float16* __restrict__ tokens,
        const _Float16* __restrict__ wT,     // [k][d] f16
        const int*      __restrict__ sel,
        const float*    __restrict__ conv_b,
        const float*    __restrict__ pos_emb,
        float*          __restrict__ out) {
    const int b    = blockIdx.x;
    const int cb   = blockIdx.y;                 // 128-col block, 0..5
    const int t    = threadIdx.x;
    const int lane = t & 31;
    const int wave = t >> 5;
    const int half = lane >> 4;                  // 0 or 1
    const int ln   = lane & 15;

    __shared__ _Float16 sA[2][VWT_ROWS * VWT_LDA]; // gathered A tiles, 112 x 32
    __shared__ _Float16 sBt[2][32 * VWT_LDB];      // B tiles, K-major 32 x 128
    __shared__ int s_sel[VWT_ROWS];

    if (t < VWT_ROWS) s_sel[t] = (t < VWT_TOPK) ? sel[b * VWT_TOPK + t] : 0;
    __syncthreads();

    // Hoisted, loop-invariant staging sources.
    const int arow = t >> 1, aseg = t & 1;        // 224 threads x 16 halves (A)
    const _Float16* srcA = tokens;                // dummy init
    if (t < 224)
        srcA = tokens + (size_t)b * VWT_NPATCH * VWT_HIDDEN
             + (size_t)s_sel[arow] * VWT_HIDDEN + aseg * 16;

    const int d0 = cb * 128;
#if !VWT_USE_TDM
    const int bkr = t >> 3, bseg = t & 7;         // 256 threads x 16 halves (B)
    const _Float16* srcB = wT + (size_t)bkr * VWT_HIDDEN + d0 + bseg * 16;
#endif

    // ---- prologue: stage K-step 0 into buffer 0 ----
    if (t < 224) {
        uint4 a0 = *(const uint4*)(srcA);
        uint4 a1 = *(const uint4*)(srcA + 8);
        _Float16* dst = &sA[0][arow * VWT_LDA + aseg * 16];
        *(uint4*)(dst)     = a0;
        *(uint4*)(dst + 8) = a1;
    }
#if VWT_USE_TDM
    if (wave == 0) {
        vwt_tdm_load_b(wT + d0, &sBt[0][0]);
        __builtin_amdgcn_s_wait_tensorcnt((short)0);
    }
#else
    {
        uint4 b0 = *(const uint4*)(srcB);
        uint4 b1 = *(const uint4*)(srcB + 8);
        _Float16* dst = &sBt[0][bkr * VWT_LDB + bseg * 16];
        *(uint4*)(dst)     = b0;
        *(uint4*)(dst + 8) = b1;
    }
#endif
    __syncthreads();

    v8f acc[7] = {};

    for (int k0 = 0; k0 < VWT_HIDDEN; k0 += 32) {
        const int  buf  = (k0 >> 5) & 1;
        const bool more = (k0 + 32 < VWT_HIDDEN);

        // ---- issue next-tile loads before compute (overlap with WMMA) ----
        uint4 a0{}, a1{};
#if !VWT_USE_TDM
        uint4 b0{}, b1{};
#endif
        if (more) {
            if (t < 224) {
                a0 = *(const uint4*)(srcA + k0 + 32);
                a1 = *(const uint4*)(srcA + k0 + 32 + 8);
            }
#if VWT_USE_TDM
            if (wave == 0)
                vwt_tdm_load_b(wT + (size_t)(k0 + 32) * VWT_HIDDEN + d0,
                               &sBt[buf ^ 1][0]);
#else
            b0 = *(const uint4*)(srcB + (size_t)(k0 + 32) * VWT_HIDDEN);
            b1 = *(const uint4*)(srcB + (size_t)(k0 + 32) * VWT_HIDDEN + 8);
            __builtin_prefetch(srcB + (size_t)(k0 + 64) * VWT_HIDDEN, 0, 1);
#endif
        }

        // ---- B fragment: lane N = ln ; K = 8*half + e, 16 + 8*half + e ----
        v16h bfrag;
        {
            const _Float16* pb = &sBt[buf][(wave << 4) + ln];
#pragma unroll
            for (int e = 0; e < 8; ++e) {
                bfrag[e]     = pb[(size_t)(half * 8 + e) * VWT_LDB];
                bfrag[e + 8] = pb[(size_t)(16 + half * 8 + e) * VWT_LDB];
            }
        }
        // ---- load all 7 A fragments first, then 7 back-to-back WMMAs ----
        v8h alo[7], ahi[7];
#pragma unroll
        for (int mt = 0; mt < 7; ++mt) {
            const _Float16* pa = &sA[buf][(size_t)(mt * 16 + ln) * VWT_LDA + half * 8];
            alo[mt] = *(const v8h*)(pa);          // K = 8*half + 0..7
            ahi[mt] = *(const v8h*)(pa + 16);     // K = 16 + 8*half + 0..7
        }
#pragma unroll
        for (int mt = 0; mt < 7; ++mt) {
            v16h afrag;
#pragma unroll
            for (int e = 0; e < 8; ++e) { afrag[e] = alo[mt][e]; afrag[e + 8] = ahi[mt][e]; }
            acc[mt] = __builtin_amdgcn_wmma_f32_16x16x32_f16(
                false, afrag, false, bfrag, (short)0, acc[mt], false, false);
        }

        // ---- commit next tile into the other buffer ----
        if (more) {
            if (t < 224) {
                _Float16* dst = &sA[buf ^ 1][arow * VWT_LDA + aseg * 16];
                *(uint4*)(dst)     = a0;
                *(uint4*)(dst + 8) = a1;
            }
#if VWT_USE_TDM
            if (wave == 0) __builtin_amdgcn_s_wait_tensorcnt((short)0);
#else
            _Float16* dst = &sBt[buf ^ 1][bkr * VWT_LDB + bseg * 16];
            *(uint4*)(dst)     = b0;
            *(uint4*)(dst + 8) = b1;
#endif
        }
        __syncthreads();
    }

    // ---- epilogue: + conv_b + pos_emb[p+1], scatter to out rows 1..98 ----
    const int   d_lane = d0 + (wave << 4) + ln;
    const float bias   = conv_b[d_lane];
#pragma unroll
    for (int mt = 0; mt < 7; ++mt) {
#pragma unroll
        for (int r = 0; r < 8; ++r) {
            int m = mt * 16 + r + (half << 3);    // D layout: M = r + 8*half
            if (m < VWT_TOPK) {
                int p = s_sel[m];
                out[((size_t)b * 99 + 1 + m) * VWT_HIDDEN + d_lane] =
                    acc[mt][r] + bias + pos_emb[(size_t)(p + 1) * VWT_HIDDEN + d_lane];
            }
        }
    }
}

// ---------------------------------------------------------------------------
// Host launcher
// ---------------------------------------------------------------------------
extern "C" void kernel_launch(void* const* d_in, const int* in_sizes, int n_in,
                              void* d_out, int out_size, void* d_ws, size_t ws_size,
                              hipStream_t stream) {
    const float* pix    = (const float*)d_in[0];   // [256,3,224,224]
    const float* conv_w = (const float*)d_in[1];   // [768,768] after flatten
    const float* conv_b = (const float*)d_in[2];   // [768]
    const float* cls    = (const float*)d_in[3];   // [768]
    const float* pos    = (const float*)d_in[4];   // [197,768]
    float* out = (float*)d_out;                    // [256,99,768]

    // Workspace layout (static, ~78.6 MB total)
    char* ws = (char*)d_ws;
    const size_t TOK_BYTES = (size_t)VWT_B * VWT_NPATCH * VWT_HIDDEN * sizeof(_Float16); // 77,070,336
    const size_t WT_BYTES  = (size_t)VWT_HIDDEN * VWT_HIDDEN * sizeof(_Float16);         //  1,179,648
    const size_t SC_BYTES  = (size_t)VWT_B * VWT_NPATCH * sizeof(float);                 //    200,704
    _Float16* tokens = (_Float16*)(ws);
    _Float16* wT     = (_Float16*)(ws + TOK_BYTES);
    float*    scores = (float*)   (ws + TOK_BYTES + WT_BYTES);
    int*      sel    = (int*)     (ws + TOK_BYTES + WT_BYTES + SC_BYTES);

    // 0) weight transpose+cast (589824 elems / 256)
    vwt_wprep<<<2304, 256, 0, stream>>>(conv_w, wT);
    // 1) score + tokenize: 256*196 waves, 8 waves/block
    vwt_score_token<<<(VWT_B * VWT_NPATCH) / 8, 256, 0, stream>>>(pix, tokens, scores);
    // 2) per-image top-98
    vwt_topk<<<VWT_B, 256, 0, stream>>>(scores, sel);
    // 3) CLS row
    vwt_cls_row<<<dim3(VWT_B, 3), 256, 0, stream>>>(cls, pos, out);
    // 4) selected-patch WMMA GEMM (double-buffered LDS, TDM-staged B tiles)
    vwt_gemm<<<dim3(VWT_B, 6), 256, 0, stream>>>(tokens, wT, sel, conv_b, pos, out);
}
